// LocalNerfTransformer_29076928594409
// MI455X (gfx1250) — compile-verified, matrix operand order checked by reading, not run on hardware
//
#include <hip/hip_runtime.h>

// ---------------------------------------------------------------------------
// Fused windowed-attention NeRF transformer for MI455X (gfx1250, wave32).
// One workgroup (256 threads = 8 wave32) per batch element; the entire
// 192x256 fp32 residual stream lives in LDS (192KB of the 320KB WGP LDS).
// All GEMMs use v_wmma_f32_16x16x32_bf16 with pre-packed bf16 B operands.
// ---------------------------------------------------------------------------

typedef __bf16 bf16_t;
typedef __attribute__((ext_vector_type(16))) __bf16 v16bf;
typedef __attribute__((ext_vector_type(8)))  __bf16 v8bf;
typedef __attribute__((ext_vector_type(8)))  float  v8f;

#define NTOK   192
#define DMODEL 256
#define NMT    12        // 192/16 token tiles
#define SHIFTV 4

// packed-weight offsets (in bf16 elements) inside d_ws
#define OFF_WIN      0                       // w_in   Kp=64  N=256 -> 16384
#define LAYER_STRIDE 475136                  // 6*65536 + 81920
#define OFF_LAYER(l) (16384 + (l)*LAYER_STRIDE)
#define OFF_Q(l)  (OFF_LAYER(l) + 0)
#define OFF_K(l)  (OFF_LAYER(l) + 65536)
#define OFF_V(l)  (OFF_LAYER(l) + 131072)
#define OFF_O(l)  (OFF_LAYER(l) + 196608)
#define OFF_M1(l) (OFF_LAYER(l) + 262144)
#define OFF_M2(l) (OFF_LAYER(l) + 327680)
#define OFF_SK(l) (OFF_LAYER(l) + 393216)    // Kp=320 N=256 -> 81920
#define OFF_FEAT  (16384 + 2*LAYER_STRIDE)   // 966656
#define OFF_RGB1  (OFF_FEAT + 65536)         // Kp=288 N=128 -> 36864
#define WS_TOTAL_HALF (OFF_RGB1 + 36864)     // 1069056 halves ~= 2.04 MB

// ---------------------------------------------------------------------------
// Weight pre-pack: fp32 row-major [K,N] -> bf16 WMMA B-operand tiles.
// Layout: dst[(((nt*KC)+kc)*32 + lane)*16 + e], mirroring the documented
// 16-bit A-matrix 16x32 lane layout with M->N.  Supports a two-segment K
// mapping with zero padding (for the 63->64 and concat-padded K dims).
// ---------------------------------------------------------------------------
__global__ void nerf_pack_b(const float* __restrict__ src, bf16_t* __restrict__ dst,
                            int N, int K0, int K0p, int K1, int Kp, int total) {
  int idx = blockIdx.x * 256 + threadIdx.x;
  if (idx >= total) return;
  int e    = idx & 15;
  int lane = (idx >> 4) & 31;
  int t    = idx >> 9;
  int KC   = Kp >> 5;
  int kc   = t % KC;
  int nt   = t / KC;
  int n    = nt * 16 + (lane & 15);
  int base = (lane >> 4) ? 8 : 0;
  int kin  = (e < 8) ? (base + e) : (base + 16 + (e - 8));
  int pk   = kc * 32 + kin;
  float v = 0.0f;
  if (pk < K0)                          v = src[pk * N + n];
  else if (pk >= K0p && pk < K0p + K1)  v = src[(K0 + pk - K0p) * N + n];
  dst[idx] = (bf16_t)v;
}

// ---------------------------------------------------------------------------
// device helpers
// ---------------------------------------------------------------------------
__device__ __forceinline__ v16bf load_a_bf16(const bf16_t* base, int stride, int k0, int lane) {
  // 16-bit A-matrix 16x32 layout: lanes 0-15 row M=lane, K {0..7,16..23};
  // lanes 16-31 row M=lane-16, K {8..15,24..31}.
  int lr = lane & 15, hi = lane >> 4;
  const bf16_t* p = base + lr * stride + k0 + hi * 8;
  v8bf a = *(const v8bf*)(p);
  v8bf b = *(const v8bf*)(p + 16);
  return __builtin_shufflevector(a, b, 0,1,2,3,4,5,6,7,8,9,10,11,12,13,14,15);
}

__device__ __forceinline__ v16bf load_a_f32(const float* base, int stride, int k0, int lane) {
  int lr = lane & 15, hi = lane >> 4;
  const float* p = base + lr * stride + k0 + hi * 8;
  v16bf A;
#pragma unroll
  for (int i = 0; i < 8; ++i) { A[i] = (__bf16)p[i]; A[i + 8] = (__bf16)p[16 + i]; }
  return A;
}

__device__ __forceinline__ v16bf load_b_pk(const bf16_t* w, int KC, int nt, int kc, int lane) {
  return *(const v16bf*)(w + ((size_t)(nt * KC + kc) * 32 + (size_t)lane) * 16);
}

__device__ __forceinline__ v8f wmma_bf16(v16bf a, v16bf b, v8f c) {
  return __builtin_amdgcn_wmma_f32_16x16x32_bf16(false, a, false, b, (short)0, c, false, false);
}

__device__ __forceinline__ float wave_sum(float v) {
#pragma unroll
  for (int m = 16; m; m >>= 1) v += __shfl_xor(v, m, 32);
  return v;
}

__device__ __forceinline__ float gelu_tanh(float x) {
  float x3 = x * x * x;
  return 0.5f * x * (1.0f + tanhf(0.7978845608028654f * (x + 0.044715f * x3)));
}

// LayerNorm one 256-wide row with one wave; write bf16 result.
__device__ __forceinline__ void ln_row(const float* x, const float* g, const float* b,
                                       bf16_t* dst, int lane) {
  float v[8];
  float s = 0.0f;
#pragma unroll
  for (int j = 0; j < 8; ++j) { v[j] = x[lane + 32 * j]; s += v[j]; }
  float mean = wave_sum(s) * (1.0f / 256.0f);
  float var = 0.0f;
#pragma unroll
  for (int j = 0; j < 8; ++j) { float d = v[j] - mean; var += d * d; }
  var = wave_sum(var) * (1.0f / 256.0f);
  float rs = rsqrtf(var + 1e-6f);
#pragma unroll
  for (int j = 0; j < 8; ++j) {
    int c = lane + 32 * j;
    dst[c] = (bf16_t)((v[j] - mean) * rs * g[c] + b[c]);
  }
}

struct KParams {
  const float* pts;    const float* views;
  const float* b_in;   const float* ln_in_s; const float* ln_in_b;
  const float* ln1_s;  const float* ln1_b;
  const float* bq;     const float* bk;      const float* bv;     const float* bo;
  const float* ln2_s;  const float* ln2_b;
  const float* b_mlp1; const float* b_mlp2;  const float* b_skip;
  const float* ln_f_s; const float* ln_f_b;
  const float* w_alpha; const float* b_alpha; const float* b_feat;
  const float* b_rgb1;  const float* w_rgb2;  const float* b_rgb2;
  const bf16_t* pk;
  float* out;
};

// ---------------------------------------------------------------------------
// main fused kernel: 1 block per batch element
// ---------------------------------------------------------------------------
__global__ __launch_bounds__(256, 1)
void nerf_fused_kernel(KParams p) {
  __shared__ float  X[NTOK * DMODEL];     // residual stream, fp32 (192 KB)
  __shared__ bf16_t P[NTOK * 64];         // input_pts padded 63->64 (24 KB)
  __shared__ bf16_t Yt[16 * DMODEL];      // LN output tile
  __shared__ bf16_t Qt[16 * DMODEL];      // also reused as rgb-hidden
  __shared__ bf16_t Kt[16 * DMODEL];
  __shared__ bf16_t Vt[16 * DMODEL];
  __shared__ bf16_t Ht[16 * DMODEL];      // attn-out / mlp-hidden / feature
  __shared__ float  Pw[8 * 64];           // per-wave softmax probs
  __shared__ bf16_t Vw[32];               // views padded 27->32

  const int tid  = threadIdx.x;
  const int lane = tid & 31;
  const int wave = tid >> 5;
  const int blk  = blockIdx.x;

  // ---- stage A: stage inputs into LDS -------------------------------------
  for (int i = tid; i < NTOK * 64; i += 256) {
    int r = i >> 6, c = i & 63;
    P[i] = (bf16_t)((c < 63) ? p.pts[((size_t)blk * NTOK + r) * 63 + c] : 0.0f);
  }
  if (tid < 32) Vw[tid] = (bf16_t)((tid < 27) ? p.views[blk * 27 + tid] : 0.0f);
  __syncthreads();

  // ---- stage B: x = pts @ w_in + b_in -------------------------------------
  {
    const bf16_t* wpk = p.pk + OFF_WIN;
    for (int mt = 0; mt < NMT; ++mt) {
      for (int tt = 0; tt < 2; ++tt) {
        int nt = wave * 2 + tt;
        v8f acc = {0.f,0.f,0.f,0.f,0.f,0.f,0.f,0.f};
#pragma unroll
        for (int kc = 0; kc < 2; ++kc)
          acc = wmma_bf16(load_a_bf16(P + mt * 16 * 64, 64, kc * 32, lane),
                          load_b_pk(wpk, 2, nt, kc, lane), acc);
        int n = nt * 16 + (lane & 15), hi = lane >> 4;
        float bi = p.b_in[n];
#pragma unroll
        for (int r = 0; r < 8; ++r) X[(mt * 16 + r + hi * 8) * DMODEL + n] = acc[r] + bi;
      }
    }
  }
  __syncthreads();

  // ---- LN_in + sinusoidal positional embedding (in place) -----------------
  for (int r = wave; r < NTOK; r += 8) {
    float* xr = X + r * DMODEL;
    float v[8];
    float s = 0.0f;
#pragma unroll
    for (int j = 0; j < 8; ++j) { v[j] = xr[lane + 32 * j]; s += v[j]; }
    float mean = wave_sum(s) * (1.0f / 256.0f);
    float var = 0.0f;
#pragma unroll
    for (int j = 0; j < 8; ++j) { float d = v[j] - mean; var += d * d; }
    var = wave_sum(var) * (1.0f / 256.0f);
    float rs = rsqrtf(var + 1e-6f);
#pragma unroll
    for (int j = 0; j < 8; ++j) {
      int c = lane + 32 * j;
      float val = (v[j] - mean) * rs * p.ln_in_s[c] + p.ln_in_b[c];
      float freq = __expf(-(float)(2 * (c >> 1)) * (1.0f / 256.0f) * 9.210340371976184f);
      float ang = (float)r * freq;
      val += (c & 1) ? __cosf(ang) : __sinf(ang);
      xr[c] = val;
    }
  }
  __syncthreads();

  // ---- transformer layers -------------------------------------------------
  for (int l = 0; l < 2; ++l) {
    const int shift = (l == 1) ? SHIFTV : 0;
    const bf16_t *pq = p.pk + OFF_Q(l), *pkk = p.pk + OFF_K(l), *pv = p.pk + OFF_V(l);
    const bf16_t *po = p.pk + OFF_O(l), *pm1 = p.pk + OFF_M1(l), *pm2 = p.pk + OFF_M2(l);
    const bf16_t *psk = p.pk + OFF_SK(l);
    const float *bq_l = p.bq + l * 256, *bk_l = p.bk + l * 256, *bv_l = p.bv + l * 256;
    const float *bo_l = p.bo + l * 256, *b1_l = p.b_mlp1 + l * 256, *b2_l = p.b_mlp2 + l * 256;
    const float *bsk_l = p.b_skip + l * 256;

    // ===== windowed attention, 16-token tiles =====
    for (int mt = 0; mt < NMT; ++mt) {
#pragma unroll
      for (int rr = 0; rr < 2; ++rr) {
        int r = wave + rr * 8;
        int src = (mt * 16 + r + shift) % NTOK;       // roll(y, -shift)
        ln_row(X + src * DMODEL, p.ln1_s + l * 256, p.ln1_b + l * 256, Yt + r * DMODEL, lane);
      }
      __syncthreads();

      // QKV: 3 matrices x 16 n-tiles = 48 tiles, 6 per wave
      for (int t = wave * 6; t < wave * 6 + 6; ++t) {
        int which = t >> 4, nt = t & 15;
        const bf16_t* wmat = (which == 0) ? pq : (which == 1) ? pkk : pv;
        const float*  bias = (which == 0) ? bq_l : (which == 1) ? bk_l : bv_l;
        bf16_t*       dstb = (which == 0) ? Qt : (which == 1) ? Kt : Vt;
        v8f acc = {0.f,0.f,0.f,0.f,0.f,0.f,0.f,0.f};
#pragma unroll
        for (int kc = 0; kc < 8; ++kc)
          acc = wmma_bf16(load_a_bf16(Yt, DMODEL, kc * 32, lane),
                          load_b_pk(wmat, 8, nt, kc, lane), acc);
        int n = nt * 16 + (lane & 15), hi = lane >> 4;
        float bi = bias[n];
#pragma unroll
        for (int r = 0; r < 8; ++r) dstb[(r + hi * 8) * DMODEL + n] = (bf16_t)(acc[r] + bi);
      }
      __syncthreads();

      // attention within the 2 windows x 8 heads of this tile (2 units/wave)
      for (int u = wave * 2; u < wave * 2 + 2; ++u) {
        int j = u >> 3, h = u & 7;
        int qi = lane & 7, kb = lane >> 3;            // kb in 0..3
        const bf16_t* qrow = Qt + (j * 8 + qi) * DMODEL + h * 32;
        const bf16_t* k0r  = Kt + (j * 8 + kb) * DMODEL + h * 32;
        const bf16_t* k1r  = Kt + (j * 8 + kb + 4) * DMODEL + h * 32;
        float s0 = 0.f, s1 = 0.f;
#pragma unroll
        for (int d = 0; d < 32; ++d) {
          float qd = (float)qrow[d];
          s0 += qd * (float)k0r[d];
          s1 += qd * (float)k1r[d];
        }
        const float scale = 0.17677669529663687f;     // 1/sqrt(32)
        s0 *= scale; s1 *= scale;
        float mx = fmaxf(s0, s1);
        mx = fmaxf(mx, __shfl_xor(mx, 8, 32));
        mx = fmaxf(mx, __shfl_xor(mx, 16, 32));
        float p0 = __expf(s0 - mx), p1 = __expf(s1 - mx);
        float sm = p0 + p1;
        sm += __shfl_xor(sm, 8, 32);
        sm += __shfl_xor(sm, 16, 32);
        float inv = 1.0f / sm;
        Pw[wave * 64 + qi * 8 + kb]     = p0 * inv;   // wave-local LDS scratch
        Pw[wave * 64 + qi * 8 + kb + 4] = p1 * inv;
        // o[:, d=lane] = P @ V
#pragma unroll
        for (int q2 = 0; q2 < 8; ++q2) {
          float o = 0.f;
#pragma unroll
          for (int k2 = 0; k2 < 8; ++k2)
            o += Pw[wave * 64 + q2 * 8 + k2] * (float)Vt[(j * 8 + k2) * DMODEL + h * 32 + lane];
          Ht[(j * 8 + q2) * DMODEL + h * 32 + lane] = (bf16_t)o;
        }
      }
      __syncthreads();

      // Wo projection + residual add (scatter back through the roll)
      for (int tt = 0; tt < 2; ++tt) {
        int nt = wave * 2 + tt;
        v8f acc = {0.f,0.f,0.f,0.f,0.f,0.f,0.f,0.f};
#pragma unroll
        for (int kc = 0; kc < 8; ++kc)
          acc = wmma_bf16(load_a_bf16(Ht, DMODEL, kc * 32, lane),
                          load_b_pk(po, 8, nt, kc, lane), acc);
        int n = nt * 16 + (lane & 15), hi = lane >> 4;
        float bi = bo_l[n];
#pragma unroll
        for (int r = 0; r < 8; ++r) {
          int dst = (mt * 16 + r + hi * 8 + shift) % NTOK;
          X[dst * DMODEL + n] += acc[r] + bi;
        }
      }
      __syncthreads();
    }

    // ===== MLP =====
    for (int mt = 0; mt < NMT; ++mt) {
#pragma unroll
      for (int rr = 0; rr < 2; ++rr) {
        int r = wave + rr * 8;
        ln_row(X + (mt * 16 + r) * DMODEL, p.ln2_s + l * 256, p.ln2_b + l * 256,
               Yt + r * DMODEL, lane);
      }
      __syncthreads();
      for (int tt = 0; tt < 2; ++tt) {
        int nt = wave * 2 + tt;
        v8f acc = {0.f,0.f,0.f,0.f,0.f,0.f,0.f,0.f};
#pragma unroll
        for (int kc = 0; kc < 8; ++kc)
          acc = wmma_bf16(load_a_bf16(Yt, DMODEL, kc * 32, lane),
                          load_b_pk(pm1, 8, nt, kc, lane), acc);
        int n = nt * 16 + (lane & 15), hi = lane >> 4;
        float bi = b1_l[n];
#pragma unroll
        for (int r = 0; r < 8; ++r)
          Ht[(r + hi * 8) * DMODEL + n] = (bf16_t)gelu_tanh(acc[r] + bi);
      }
      __syncthreads();
      for (int tt = 0; tt < 2; ++tt) {
        int nt = wave * 2 + tt;
        v8f acc = {0.f,0.f,0.f,0.f,0.f,0.f,0.f,0.f};
#pragma unroll
        for (int kc = 0; kc < 8; ++kc)
          acc = wmma_bf16(load_a_bf16(Ht, DMODEL, kc * 32, lane),
                          load_b_pk(pm2, 8, nt, kc, lane), acc);
        int n = nt * 16 + (lane & 15), hi = lane >> 4;
        float bi = b2_l[n];
#pragma unroll
        for (int r = 0; r < 8; ++r) X[(mt * 16 + r + hi * 8) * DMODEL + n] += acc[r] + bi;
      }
      __syncthreads();
    }

    // ===== skip projection: x = gelu([pts | x] @ w_skip + b) (in place) =====
    for (int mt = 0; mt < NMT; ++mt) {
      v8f acc0 = {0.f,0.f,0.f,0.f,0.f,0.f,0.f,0.f};
      v8f acc1 = acc0;
#pragma unroll
      for (int kc = 0; kc < 10; ++kc) {
        v16bf A = (kc < 2) ? load_a_bf16(P + mt * 16 * 64, 64, kc * 32, lane)
                           : load_a_f32(X + mt * 16 * DMODEL, DMODEL, (kc - 2) * 32, lane);
        acc0 = wmma_bf16(A, load_b_pk(psk, 10, wave * 2 + 0, kc, lane), acc0);
        acc1 = wmma_bf16(A, load_b_pk(psk, 10, wave * 2 + 1, kc, lane), acc1);
      }
      __syncthreads();                                // all reads of X done
      int lr = lane & 15, hi = lane >> 4;
      for (int tt = 0; tt < 2; ++tt) {
        int n = (wave * 2 + tt) * 16 + lr;
        float bi = bsk_l[n];
        v8f& acc = tt ? acc1 : acc0;
#pragma unroll
        for (int r = 0; r < 8; ++r)
          X[(mt * 16 + r + hi * 8) * DMODEL + n] = gelu_tanh(acc[r] + bi);
      }
      __syncthreads();
    }
  }

  // ---- head: LN_f, alpha, feature, rgb ------------------------------------
  {
    const bf16_t* pfe = p.pk + OFF_FEAT;
    const bf16_t* pr1 = p.pk + OFF_RGB1;
    float* out_rgb   = p.out;
    float* out_alpha = p.out + (size_t)1024 * NTOK * 3;
    bf16_t* Rt = Qt;                                  // reuse as 16x128 rgb-hidden

    for (int mt = 0; mt < NMT; ++mt) {
#pragma unroll
      for (int rr = 0; rr < 2; ++rr) {
        int r = wave + rr * 8;
        ln_row(X + (mt * 16 + r) * DMODEL, p.ln_f_s, p.ln_f_b, Yt + r * DMODEL, lane);
      }
      __syncthreads();

      // feature = h @ w_feat + b_feat  -> Ht (bf16)
      for (int tt = 0; tt < 2; ++tt) {
        int nt = wave * 2 + tt;
        v8f acc = {0.f,0.f,0.f,0.f,0.f,0.f,0.f,0.f};
#pragma unroll
        for (int kc = 0; kc < 8; ++kc)
          acc = wmma_bf16(load_a_bf16(Yt, DMODEL, kc * 32, lane),
                          load_b_pk(pfe, 8, nt, kc, lane), acc);
        int n = nt * 16 + (lane & 15), hi = lane >> 4;
        float bi = p.b_feat[n];
#pragma unroll
        for (int r = 0; r < 8; ++r) Ht[(r + hi * 8) * DMODEL + n] = (bf16_t)(acc[r] + bi);
      }
      // alpha = h @ w_alpha + b_alpha
      if (tid < 16) {
        float dot = 0.f;
        for (int c = 0; c < 256; ++c) dot += (float)Yt[tid * DMODEL + c] * p.w_alpha[c];
        out_alpha[(size_t)blk * NTOK + mt * 16 + tid] = dot + p.b_alpha[0];
      }
      __syncthreads();

      // rgb hidden = gelu([feature | views] @ w_rgb1 + b)  (N=128, 1 tile/wave)
      {
        v8f acc = {0.f,0.f,0.f,0.f,0.f,0.f,0.f,0.f};
#pragma unroll
        for (int kc = 0; kc < 9; ++kc) {
          v16bf A = (kc < 8) ? load_a_bf16(Ht, DMODEL, kc * 32, lane)
                             : load_a_bf16(Vw, 0, 0, lane);    // broadcast views row
          acc = wmma_bf16(A, load_b_pk(pr1, 9, wave, kc, lane), acc);
        }
        int n = wave * 16 + (lane & 15), hi = lane >> 4;
        float bi = p.b_rgb1[n];
#pragma unroll
        for (int r = 0; r < 8; ++r)
          Rt[(r + hi * 8) * 128 + n] = (bf16_t)gelu_tanh(acc[r] + bi);
      }
      __syncthreads();

      // rgb = hidden @ w_rgb2 + b_rgb2   (N=3, plain VALU dot products)
      if (tid < 48) {
        int row = tid / 3, c = tid % 3;
        float dot = p.b_rgb2[c];
        for (int k = 0; k < 128; ++k)
          dot += (float)Rt[row * 128 + k] * p.w_rgb2[k * 3 + c];
        out_rgb[((size_t)blk * NTOK + mt * 16 + row) * 3 + c] = dot;
      }
      __syncthreads();
    }
  }
}

// ---------------------------------------------------------------------------
// host launcher
// ---------------------------------------------------------------------------
extern "C" void kernel_launch(void* const* d_in, const int* in_sizes, int n_in,
                              void* d_out, int out_size, void* d_ws, size_t ws_size,
                              hipStream_t stream) {
  const float* pts     = (const float*)d_in[0];
  const float* views   = (const float*)d_in[1];
  const float* w_in    = (const float*)d_in[2];
  const float* b_in    = (const float*)d_in[3];
  const float* ln_in_s = (const float*)d_in[4];
  const float* ln_in_b = (const float*)d_in[5];
  const float* ln1_s   = (const float*)d_in[6];
  const float* ln1_b   = (const float*)d_in[7];
  const float* wq      = (const float*)d_in[8];
  const float* bq      = (const float*)d_in[9];
  const float* wk      = (const float*)d_in[10];
  const float* bk      = (const float*)d_in[11];
  const float* wv      = (const float*)d_in[12];
  const float* bv      = (const float*)d_in[13];
  const float* wo      = (const float*)d_in[14];
  const float* bo      = (const float*)d_in[15];
  const float* ln2_s   = (const float*)d_in[16];
  const float* ln2_b   = (const float*)d_in[17];
  const float* w_mlp1  = (const float*)d_in[18];
  const float* b_mlp1  = (const float*)d_in[19];
  const float* w_mlp2  = (const float*)d_in[20];
  const float* b_mlp2  = (const float*)d_in[21];
  const float* w_skip  = (const float*)d_in[22];
  const float* b_skip  = (const float*)d_in[23];
  const float* ln_f_s  = (const float*)d_in[24];
  const float* ln_f_b  = (const float*)d_in[25];
  const float* w_alpha = (const float*)d_in[26];
  const float* b_alpha = (const float*)d_in[27];
  const float* w_feat  = (const float*)d_in[28];
  const float* b_feat  = (const float*)d_in[29];
  const float* w_rgb1  = (const float*)d_in[30];
  const float* b_rgb1  = (const float*)d_in[31];
  const float* w_rgb2  = (const float*)d_in[32];
  const float* b_rgb2  = (const float*)d_in[33];
  (void)in_sizes; (void)n_in; (void)out_size; (void)ws_size;

  bf16_t* ws = (bf16_t*)d_ws;
  auto pack = [&](const float* src, size_t off, int N, int K0, int K0p, int K1, int Kp) {
    int total = Kp * N;
    nerf_pack_b<<<(total + 255) / 256, 256, 0, stream>>>(src, ws + off, N, K0, K0p, K1, Kp, total);
  };

  pack(w_in, OFF_WIN, 256, 63, 64, 0, 64);
  for (int l = 0; l < 2; ++l) {
    pack(wq     + (size_t)l * 65536,     OFF_Q(l),  256, 256, 256, 0,   256);
    pack(wk     + (size_t)l * 65536,     OFF_K(l),  256, 256, 256, 0,   256);
    pack(wv     + (size_t)l * 65536,     OFF_V(l),  256, 256, 256, 0,   256);
    pack(wo     + (size_t)l * 65536,     OFF_O(l),  256, 256, 256, 0,   256);
    pack(w_mlp1 + (size_t)l * 65536,     OFF_M1(l), 256, 256, 256, 0,   256);
    pack(w_mlp2 + (size_t)l * 65536,     OFF_M2(l), 256, 256, 256, 0,   256);
    pack(w_skip + (size_t)l * 319 * 256, OFF_SK(l), 256, 63,  64,  256, 320);
  }
  pack(w_feat, OFF_FEAT, 256, 256, 256, 0, 256);
  pack(w_rgb1, OFF_RGB1, 128, 256, 256, 27, 288);

  KParams p;
  p.pts = pts; p.views = views;
  p.b_in = b_in; p.ln_in_s = ln_in_s; p.ln_in_b = ln_in_b;
  p.ln1_s = ln1_s; p.ln1_b = ln1_b;
  p.bq = bq; p.bk = bk; p.bv = bv; p.bo = bo;
  p.ln2_s = ln2_s; p.ln2_b = ln2_b;
  p.b_mlp1 = b_mlp1; p.b_mlp2 = b_mlp2; p.b_skip = b_skip;
  p.ln_f_s = ln_f_s; p.ln_f_b = ln_f_b;
  p.w_alpha = w_alpha; p.b_alpha = b_alpha; p.b_feat = b_feat;
  p.b_rgb1 = b_rgb1; p.w_rgb2 = w_rgb2; p.b_rgb2 = b_rgb2;
  p.pk = ws;
  p.out = (float*)d_out;

  nerf_fused_kernel<<<1024, 256, 0, stream>>>(p);
}
